// SwinTransformerBlock_48524540510954
// MI455X (gfx1250) — matrix operand order repaired and checked by reading, not compiled, not run on hardware
//
#include <hip/hip_runtime.h>

typedef unsigned short u16;
typedef __attribute__((ext_vector_type(16))) __bf16 v16bf;
typedef __attribute__((ext_vector_type(8)))  float  v8f;

// ---------------- problem constants (compile-time) ----------------
constexpr int   SHIFTC = 3;
constexpr int   HEADS  = 16;
constexpr int   CDIM   = 512;
constexpr int   HD     = 32;
constexpr int   NREF   = 100;
constexpr int   NREFP  = 128;            // padded K for ra @ ref_v
constexpr int   HIMG   = 112;
constexpr int   WIMG   = 112;
constexpr int   BATCH  = 2;
constexpr int   NTOK   = 49;             // 7*7 tokens per window
constexpr int   NTOKP  = 64;             // padded rows/cols for attention tiles
constexpr int   BWIN   = BATCH * 256;    // 512 window-batches
constexpr int   MROWS  = BWIN * NTOK;    // 25088
constexpr int   LTOK   = HIMG * WIMG;    // 12544
constexpr float SCALEQ = 0.17677669529663687f;   // 32^-0.5
constexpr float EPSLN  = 1e-5f;

// ---------------- workspace layout (bytes) ----------------
constexpr size_t OFF_XW    = 0;                          // xw_bf -> later attnout_bf -> later h_bf
constexpr size_t SZ_XW     = (size_t)MROWS * CDIM * 2;   // 25,690,112
constexpr size_t OFF_QKV   = OFF_XW + SZ_XW;             // qkv_bf (25088x1536)
constexpr size_t SZ_QKV    = (size_t)MROWS * 3 * CDIM * 2;
constexpr size_t OFF_REFQK = OFF_QKV + SZ_QKV;           // refqk f32 (200x1024)
constexpr size_t SZ_REFQK  = (size_t)200 * 1024 * 4;
constexpr size_t OFF_REFQ  = OFF_REFQK + SZ_REFQK;       // refq_bf (2*16 x 128 x 32)
constexpr size_t SZ_REFQ   = (size_t)BATCH * HEADS * NREFP * HD * 2;
constexpr size_t OFF_REFVT = OFF_REFQ + SZ_REFQ;         // refvT_bf (2*16 x 32 x 128)
constexpr size_t SZ_REFVT  = (size_t)BATCH * HEADS * HD * NREFP * 2;
constexpr size_t OFF_XREF  = OFF_REFVT + SZ_REFVT;       // xref_bf (200x512)
constexpr size_t SZ_XREF   = (size_t)200 * CDIM * 2;
constexpr size_t OFF_STATS = OFF_XREF + SZ_XREF;         // 64 f32
constexpr size_t SZ_STATS  = 256;
constexpr size_t OFF_RA    = OFF_STATS + SZ_STATS;       // ra f32 conv-layout (2,16,12544,100)
constexpr size_t SZ_RA     = (size_t)BATCH * HEADS * LTOK * NREF * 4;  // 160,563,200
constexpr size_t OFF_ATTNF = OFF_RA;                     // attn_f32 padded (512,16,64,64) = 134,217,728 (ra dead)
constexpr size_t OFF_U     = OFF_RA + SZ_RA;             // u f32 (conv temp)
constexpr size_t SZ_U      = SZ_RA;
constexpr size_t OFF_RABF  = OFF_U;                      // ra_bf (512,16,49,128) aliases dead u
constexpr size_t OFF_ATTNB = OFF_U + 104857600;          // attn_bf (512,16,49,64) after ra_bf dead
constexpr size_t OFF_H1    = OFF_U;                      // h1_bf (25088x2048) aliases dead ra_bf/attn_bf
constexpr size_t OFF_QNEW  = OFF_U + SZ_U;               // qnew_bf padded (512,16,64,32)
constexpr size_t SZ_QNEW   = (size_t)BWIN * HEADS * NTOKP * HD * 2;
constexpr size_t OFF_VT    = OFF_QNEW + SZ_QNEW;         // vT_bf (512,16,32,64)
constexpr size_t SZ_VT     = (size_t)BWIN * HEADS * HD * NTOKP * 2;
constexpr size_t OFF_WBF   = OFF_VT + SZ_VT;             // bf16 weights
constexpr size_t OFF_WQKV  = OFF_WBF;                    // 1536x512
constexpr size_t OFF_WRQK  = OFF_WQKV + (size_t)1536 * 512 * 2;   // 1024x512
constexpr size_t OFF_WPROJ = OFF_WRQK + (size_t)1024 * 512 * 2;   // 512x512
constexpr size_t OFF_WFC1  = OFF_WPROJ + (size_t)512 * 512 * 2;   // 2048x512
constexpr size_t OFF_WFC2  = OFF_WFC1 + (size_t)2048 * 512 * 2;   // 512x2048

// ---------------- helpers ----------------
__device__ inline u16 f2bf(float f) {
  unsigned int u = __builtin_bit_cast(unsigned int, f);
  unsigned int r = u + 0x7FFFu + ((u >> 16) & 1u);
  return (u16)(r >> 16);
}
__device__ inline __bf16 bfr(u16 s) { return __builtin_bit_cast(__bf16, s); }
__device__ inline float gelu_f(float x) { return 0.5f * x * (1.0f + erff(x * 0.70710678118654752f)); }

__device__ inline v8f wmma_bf(v16bf a, v16bf b, v8f c) {
  return __builtin_amdgcn_wmma_f32_16x16x32_bf16(false, a, false, b, (short)0, c, false, false);
}

// CDNA5 async global->LDS staging (ASYNCcnt-tracked, bypasses VGPRs)
__device__ inline unsigned lds_off(const void* p) { return (unsigned)(unsigned long long)p; }
__device__ inline void async_ld_b128(unsigned ldsdst, const void* gsrc) {
  asm volatile("global_load_async_to_lds_b128 %0, %1, off"
               :: "v"(ldsdst), "v"(gsrc) : "memory");
}
__device__ inline void wait_async0() {
  asm volatile("s_wait_asynccnt 0x0" ::: "memory");
}

// window row (b_,m) -> original token row; same map for gather (LN1) and scatter (proj)
__device__ inline int win_src_row(int row) {
  int m  = row % NTOK;  int b_ = row / NTOK;
  int b  = b_ >> 8;     int win = b_ & 255;
  int wh = win >> 4, ww = win & 15;
  int r = m / 7, c2 = m % 7;
  int hh = wh * 7 + r + SHIFTC;  if (hh >= HIMG) hh -= HIMG;
  int wv = ww * 7 + c2 + SHIFTC; if (wv >= WIMG) wv -= WIMG;
  return b * LTOK + hh * WIMG + wv;
}

// A fragment (16x32 bf16): lane L<16 row M=L holds K 0..7,16..23; lane L+16 holds K 8..15,24..31 (ISA 7.12.2)
__device__ inline v16bf frag_a(const u16* base, int row0, int rstride, int rmax, int lane, int kofs) {
  int m = row0 + (lane & 15); m = (m > rmax) ? rmax : m;
  int kh = lane >> 4;
  const u16* rp = base + (size_t)m * rstride + kofs;
  v16bf f;
#pragma unroll
  for (int i = 0; i < 8; ++i) {
    int kp = (i < 4) ? (8 * kh + 2 * i) : (16 + 8 * kh + 2 * (i - 4));
    f[2 * i]     = bfr(rp[kp]);
    f[2 * i + 1] = bfr(rp[kp + 1]);
  }
  return f;
}
// B fragment from B^T stored [N][K]: lane n=lane&15, K = 8*(lane>>4)+e (e<8), 16+8*(lane>>4)+(e-8) (e>=8)
__device__ inline v16bf frag_b(const u16* base, int n0, int rstride, int nmax, int lane, int kofs) {
  int n = n0 + (lane & 15); n = (n > nmax) ? nmax : n;
  int kh = lane >> 4;
  const u16* rp = base + (size_t)n * rstride + kofs + 8 * kh;
  v16bf f;
#pragma unroll
  for (int e = 0; e < 16; ++e) f[e] = bfr(rp[(e < 8) ? e : (e + 8)]);
  return f;
}

// ---------------- LayerNorm kernels ----------------
__global__ __launch_bounds__(256) void ln_win_kernel(const float* __restrict__ x,
                                                     const float* __restrict__ w,
                                                     const float* __restrict__ b,
                                                     u16* __restrict__ out) {
  int row = blockIdx.x, tid = threadIdx.x;
  const float* src = x + (size_t)win_src_row(row) * CDIM;
  float a0 = src[tid], a1 = src[tid + 256];
  __shared__ float s1[256], s2[256];
  s1[tid] = a0 + a1; s2[tid] = a0 * a0 + a1 * a1;
  __syncthreads();
  for (int s = 128; s > 0; s >>= 1) { if (tid < s) { s1[tid] += s1[tid + s]; s2[tid] += s2[tid + s]; } __syncthreads(); }
  float mu = s1[0] / (float)CDIM;
  float var = s2[0] / (float)CDIM - mu * mu;
  float rstd = rsqrtf(var + EPSLN);
  out[(size_t)row * CDIM + tid]       = f2bf((a0 - mu) * rstd * w[tid] + b[tid]);
  out[(size_t)row * CDIM + tid + 256] = f2bf((a1 - mu) * rstd * w[tid + 256] + b[tid + 256]);
}

__global__ __launch_bounds__(256) void ln_plain_kernel(const float* __restrict__ x,
                                                       const float* __restrict__ w,
                                                       const float* __restrict__ b,
                                                       u16* __restrict__ out) {
  int row = blockIdx.x, tid = threadIdx.x;
  const float* src = x + (size_t)row * CDIM;
  float a0 = src[tid], a1 = src[tid + 256];
  __shared__ float s1[256], s2[256];
  s1[tid] = a0 + a1; s2[tid] = a0 * a0 + a1 * a1;
  __syncthreads();
  for (int s = 128; s > 0; s >>= 1) { if (tid < s) { s1[tid] += s1[tid + s]; s2[tid] += s2[tid + s]; } __syncthreads(); }
  float mu = s1[0] / (float)CDIM;
  float var = s2[0] / (float)CDIM - mu * mu;
  float rstd = rsqrtf(var + EPSLN);
  out[(size_t)row * CDIM + tid]       = f2bf((a0 - mu) * rstd * w[tid] + b[tid]);
  out[(size_t)row * CDIM + tid + 256] = f2bf((a1 - mu) * rstd * w[tid + 256] + b[tid + 256]);
}

__global__ __launch_bounds__(256) void cvt_bf16_kernel(const float* __restrict__ src, u16* __restrict__ dst, int n) {
  int i = blockIdx.x * 256 + threadIdx.x;
  if (i < n) dst[i] = f2bf(src[i]);
}

// ---------------- big LDS-tiled WMMA GEMM: C = act(A @ Wb^T + bias) ----------------
// A: MxK bf16 row-major; Wb: NxK bf16 row-major (pre-converted weight (out,in)) == B^T layout.
constexpr int BM = 128, BN = 128, BK = 32, LDSS = 40;
constexpr int GM_BF16 = 0, GM_F32 = 1, GM_GELU = 2, GM_PROJ = 3, GM_ADD = 4;

template <int MODE, int M, int N, int K>
__global__ __launch_bounds__(256) void gemm_bf16_kernel(const u16* __restrict__ A,
                                                        const u16* __restrict__ Wb,
                                                        const float* __restrict__ bias,
                                                        u16* __restrict__ out_bf,
                                                        float* __restrict__ out_f,
                                                        const float* __restrict__ resid) {
  __shared__ u16 As[BM * LDSS];
  __shared__ u16 Bs[BN * LDSS];
  int tid = threadIdx.x, lane = tid & 31, wid = tid >> 5;
  int bm = blockIdx.y * BM, bn = blockIdx.x * BN;
  int wr = wid >> 2, wc = wid & 3;   // wave tile: 64 rows x 32 cols
  v8f acc[4][2];
#pragma unroll
  for (int i = 0; i < 4; ++i) { acc[i][0] = {}; acc[i][1] = {}; }

  int row0 = tid >> 2;          // 0..63 (each thread stages rows row0 and row0+64)
  int seg  = tid & 3;           // 16-byte segment within a 64B row
  unsigned a0 = lds_off(&As[(row0)      * LDSS + seg * 8]);
  unsigned a1 = lds_off(&As[(row0 + 64) * LDSS + seg * 8]);
  unsigned b0 = lds_off(&Bs[(row0)      * LDSS + seg * 8]);
  unsigned b1 = lds_off(&Bs[(row0 + 64) * LDSS + seg * 8]);

  for (int k0 = 0; k0 < K; k0 += BK) {
    __syncthreads();
    // CDNA5 async global->LDS tile staging (no VGPR round-trip)
    async_ld_b128(a0, A  + (size_t)(bm + row0)      * K + k0 + seg * 8);
    async_ld_b128(a1, A  + (size_t)(bm + row0 + 64) * K + k0 + seg * 8);
    async_ld_b128(b0, Wb + (size_t)(bn + row0)      * K + k0 + seg * 8);
    async_ld_b128(b1, Wb + (size_t)(bn + row0 + 64) * K + k0 + seg * 8);
    if (k0 + BK < K) {   // prefetch next tiles into L2
      __builtin_prefetch(A  + (size_t)(bm + row0) * K + k0 + BK + seg * 8, 0, 1);
      __builtin_prefetch(Wb + (size_t)(bn + row0) * K + k0 + BK + seg * 8, 0, 1);
    }
    wait_async0();
    __syncthreads();
    v16bf af[4];
#pragma unroll
    for (int mt = 0; mt < 4; ++mt) af[mt] = frag_a(As, wr * 64 + mt * 16, LDSS, BM - 1, lane, 0);
#pragma unroll
    for (int nt = 0; nt < 2; ++nt) {
      v16bf bf = frag_b(Bs, wc * 32 + nt * 16, LDSS, BN - 1, lane, 0);
#pragma unroll
      for (int mt = 0; mt < 4; ++mt) acc[mt][nt] = wmma_bf(af[mt], bf, acc[mt][nt]);
    }
  }
#pragma unroll
  for (int mt = 0; mt < 4; ++mt) {
#pragma unroll
    for (int nt = 0; nt < 2; ++nt) {
      int col = bn + wc * 32 + nt * 16 + (lane & 15);
      int rb0 = bm + wr * 64 + mt * 16 + 8 * (lane >> 4);
      float bv = bias[col];
#pragma unroll
      for (int j = 0; j < 8; ++j) {
        int row = rb0 + j;
        if (row >= M) continue;           // folds away when M % 128 == 0
        float v = acc[mt][nt][j] + bv;
        if constexpr (MODE == GM_BF16) {
          out_bf[(size_t)row * N + col] = f2bf(v);
        } else if constexpr (MODE == GM_F32) {
          out_f[(size_t)row * N + col] = v;
        } else if constexpr (MODE == GM_GELU) {
          out_bf[(size_t)row * N + col] = f2bf(gelu_f(v));
        } else if constexpr (MODE == GM_PROJ) {
          int orow = win_src_row(row);
          out_f[(size_t)orow * N + col] = resid[(size_t)orow * N + col] + v;
        } else if constexpr (MODE == GM_ADD) {
          out_f[(size_t)row * N + col] += v;
        }
      }
    }
  }
}

// ---------------- reference token prep ----------------
__global__ __launch_bounds__(256) void build_refs_kernel(const float* __restrict__ refqk,
                                                         const float* __restrict__ dmu,
                                                         const float* __restrict__ dls,
                                                         u16* __restrict__ refq,
                                                         u16* __restrict__ refvT) {
  int idx = blockIdx.x * 256 + threadIdx.x;     // 2*100*512
  int c = idx % CDIM;
  int r = (idx / CDIM) % NREF;
  int rb = idx / (CDIM * NREF);
  int h = c >> 5, d = c & 31;
  const float* row = refqk + ((size_t)rb * NREF + r) * 1024;
  float qv = dmu[c] + expf(dls[c]) * row[c];
  float vv = row[512 + c];
  refq[((size_t)(rb * HEADS + h) * NREFP + r) * HD + d] = f2bf(qv);
  refvT[((size_t)(rb * HEADS + h) * HD + d) * NREFP + r] = f2bf(vv);
}

// ---------------- small WMMA GEMMs ----------------
// ra = q @ ref_k^T : per (b_,h): (49x32)@(32x100); one wave per (b_,h,mtile), 7 n-tiles
__global__ __launch_bounds__(256) void ra_gemm_kernel(const u16* __restrict__ qkv,
                                                      const u16* __restrict__ refq,
                                                      float* __restrict__ ra) {
  int lane = threadIdx.x & 31;
  int gw = blockIdx.x * 8 + (threadIdx.x >> 5);
  int mt = gw & 3, h = (gw >> 2) & 15, b_ = gw >> 6;
  int rb = b_ >> 8, win = b_ & 255;
  const u16* abase = qkv + (size_t)b_ * NTOK * 1536 + h * HD;               // q
  const u16* bbase = refq + (size_t)(rb * HEADS + h) * NREFP * HD;
  v16bf af = frag_a(abase, mt * 16, 1536, NTOK - 1, lane, 0);
  float* obase = ra + ((size_t)(rb * HEADS + h) * LTOK + (size_t)win * NTOK) * NREF;
#pragma unroll
  for (int nt = 0; nt < 7; ++nt) {
    v16bf bf = frag_b(bbase, nt * 16, HD, NREFP - 1, lane, 0);
    v8f c = {};
    c = wmma_bf(af, bf, c);
    int col = nt * 16 + (lane & 15);
    int rb0 = mt * 16 + 8 * (lane >> 4);
    if (col < NREF) {
#pragma unroll
      for (int j = 0; j < 8; ++j) {
        int m = rb0 + j;
        if (m < NTOK) obase[(size_t)m * NREF + col] = c[j];
      }
    }
  }
}

// q_new = softmax(ra) @ ref_v * SCALE : (49x128pad)@(128x32); padded 64-row output, unconditional stores
__global__ __launch_bounds__(256) void qnew_gemm_kernel(const u16* __restrict__ rabf,
                                                        const u16* __restrict__ refvT,
                                                        u16* __restrict__ qnew) {
  int lane = threadIdx.x & 31;
  int gw = blockIdx.x * 8 + (threadIdx.x >> 5);
  int mt = gw & 3, h = (gw >> 2) & 15, b_ = gw >> 6;
  int rb = b_ >> 8;
  const u16* abase = rabf + (size_t)(b_ * HEADS + h) * NTOK * NREFP;
  const u16* bbase = refvT + (size_t)(rb * HEADS + h) * HD * NREFP;
  v8f acc[2]; acc[0] = {}; acc[1] = {};
#pragma unroll
  for (int kt = 0; kt < 4; ++kt) {
    v16bf af = frag_a(abase, mt * 16, NREFP, NTOK - 1, lane, kt * 32);
#pragma unroll
    for (int nt = 0; nt < 2; ++nt) {
      v16bf bf = frag_b(bbase, nt * 16, NREFP, HD - 1, lane, kt * 32);
      acc[nt] = wmma_bf(af, bf, acc[nt]);
    }
  }
  u16* obase = qnew + (size_t)(b_ * HEADS + h) * NTOKP * HD;
#pragma unroll
  for (int nt = 0; nt < 2; ++nt) {
    int col = nt * 16 + (lane & 15);
    int rb0 = mt * 16 + 8 * (lane >> 4);
#pragma unroll
    for (int j = 0; j < 8; ++j)
      obase[(size_t)(rb0 + j) * HD + col] = f2bf(acc[nt][j] * SCALEQ);
  }
}

// attn = q_new @ k^T + rpb + mask : (49x32)@(32x49); padded 64x64 logits, unconditional stores
__global__ __launch_bounds__(256) void attn_gemm_kernel(const u16* __restrict__ qnew,
                                                        const u16* __restrict__ qkv,
                                                        const float* __restrict__ rpb,
                                                        const float* __restrict__ mask,
                                                        float* __restrict__ attnf) {
  int lane = threadIdx.x & 31;
  int gw = blockIdx.x * 8 + (threadIdx.x >> 5);
  int mt = gw & 3, h = (gw >> 2) & 15, b_ = gw >> 6;
  int win = b_ & 255;
  const u16* abase = qnew + (size_t)(b_ * HEADS + h) * NTOKP * HD;
  const u16* bbase = qkv + (size_t)b_ * NTOK * 1536 + CDIM + h * HD;        // k
  v16bf af = frag_a(abase, mt * 16, HD, NTOK - 1, lane, 0);
  float* obase = attnf + (size_t)(b_ * HEADS + h) * NTOKP * NTOKP;
#pragma unroll
  for (int nt = 0; nt < 4; ++nt) {
    v16bf bf = frag_b(bbase, nt * 16, 1536, NTOK - 1, lane, 0);
    v8f c = {};
    c = wmma_bf(af, bf, c);
    int col = nt * 16 + (lane & 15);
    int rb0 = mt * 16 + 8 * (lane >> 4);
    int cc = (col > 48) ? 48 : col;
    int r2 = cc / 7, c2 = cc % 7;
#pragma unroll
    for (int j = 0; j < 8; ++j) {
      int m = rb0 + j;
      int mc = (m > 48) ? 48 : m;
      int r1 = mc / 7, c1 = mc % 7;
      int rpi = (r1 - r2 + 6) * 13 + (c1 - c2 + 6);
      float v = c[j] + rpb[rpi * HEADS + h] + mask[((size_t)win * NTOK + mc) * NTOK + cc];
      obase[(size_t)m * NTOKP + col] = v;
    }
  }
}

// out = softmax(attn) @ v : (49x64pad)@(64x32) with v^T staged
__global__ __launch_bounds__(256) void out_gemm_kernel(const u16* __restrict__ attnbf,
                                                       const u16* __restrict__ vT,
                                                       u16* __restrict__ attnout) {
  int lane = threadIdx.x & 31;
  int gw = blockIdx.x * 8 + (threadIdx.x >> 5);
  int mt = gw & 3, h = (gw >> 2) & 15, b_ = gw >> 6;
  const u16* abase = attnbf + (size_t)(b_ * HEADS + h) * NTOK * NTOKP;
  const u16* bbase = vT + (size_t)(b_ * HEADS + h) * HD * NTOKP;
  v8f acc[2]; acc[0] = {}; acc[1] = {};
#pragma unroll
  for (int kt = 0; kt < 2; ++kt) {
    v16bf af = frag_a(abase, mt * 16, NTOKP, NTOK - 1, lane, kt * 32);
#pragma unroll
    for (int nt = 0; nt < 2; ++nt) {
      v16bf bf = frag_b(bbase, nt * 16, NTOKP, HD - 1, lane, kt * 32);
      acc[nt] = wmma_bf(af, bf, acc[nt]);
    }
  }
#pragma unroll
  for (int nt = 0; nt < 2; ++nt) {
    int col = nt * 16 + (lane & 15);
    int rb0 = mt * 16 + 8 * (lane >> 4);
#pragma unroll
    for (int j = 0; j < 8; ++j) {
      int m = rb0 + j;
      if (m < NTOK) attnout[((size_t)b_ * NTOK + m) * CDIM + h * HD + col] = f2bf(acc[nt][j]);
    }
  }
}

// ---------------- conv diffusion ----------------
__global__ __launch_bounds__(64) void stats_zero_kernel(float* __restrict__ stats) {
  stats[threadIdx.x] = 0.0f;
}

__global__ __launch_bounds__(256) void conv_kernel(const float* __restrict__ ra,
                                                   const float* __restrict__ cw,
                                                   const float* __restrict__ cb,
                                                   float* __restrict__ u,
                                                   float* __restrict__ stats) {
  __shared__ float sw[HEADS * HEADS * 9];
  __shared__ float r1[256], r2[256];
  int tid = threadIdx.x;
  for (int j = tid; j < HEADS * HEADS * 9; j += 256) sw[j] = cw[j];
  __syncthreads();
  int idx = blockIdx.x * 256 + tid;                 // (rb, o, hh, wv)
  int wv = idx % NREF;
  int t1 = idx / NREF;
  int hh = t1 % LTOK;
  int t2 = t1 / LTOK;
  int o = t2 & 15, rbb = t2 >> 4;
  float acc = cb[o];
  const float* base = ra + (size_t)rbb * HEADS * LTOK * NREF;
  for (int i = 0; i < HEADS; ++i) {
    const float* pi = base + (size_t)i * LTOK * NREF;
    const float* wr = sw + (o * HEADS + i) * 9;
#pragma unroll
    for (int kh = 0; kh < 3; ++kh) {
      int h2 = hh + kh - 1;
      if (h2 < 0 || h2 >= LTOK) continue;
#pragma unroll
      for (int kw = 0; kw < 3; ++kw) {
        int w2 = wv + kw - 1;
        if (w2 < 0 || w2 >= NREF) continue;
        acc += pi[(size_t)h2 * NREF + w2] * wr[kh * 3 + kw];
      }
    }
  }
  u[idx] = acc;
  r1[tid] = acc; r2[tid] = acc * acc;
  __syncthreads();
  for (int s = 128; s > 0; s >>= 1) { if (tid < s) { r1[tid] += r1[tid + s]; r2[tid] += r2[tid + s]; } __syncthreads(); }
  if (tid == 0) {
    atomicAdd(&stats[(rbb << 4) + o], r1[0]);
    atomicAdd(&stats[32 + (rbb << 4) + o], r2[0]);
  }
}

__global__ __launch_bounds__(256) void conv_update_kernel(const float* __restrict__ u,
                                                          const float* __restrict__ stats,
                                                          float* __restrict__ ra) {
  int idx = blockIdx.x * 256 + threadIdx.x;
  int t2 = idx / (LTOK * NREF);
  int o = t2 & 15, rbb = t2 >> 4;
  float cnt = (float)LTOK * (float)NREF;
  float mu = stats[(rbb << 4) + o] / cnt;
  float var = stats[32 + (rbb << 4) + o] / cnt - mu * mu;
  float xn = (u[idx] - mu) * rsqrtf(var + EPSLN);
  ra[idx] += gelu_f(xn);
}

// ---------------- softmaxes ----------------
__global__ __launch_bounds__(128) void softmax_ra_kernel(const float* __restrict__ ra, u16* __restrict__ out) {
  int row = blockIdx.x, tid = threadIdx.x;       // row = (b_*16+h)*49+m
  int m = row % NTOK; int t = row / NTOK;
  int h = t & 15, b_ = t >> 4;
  int rb = b_ >> 8, win = b_ & 255;
  const float* src = ra + ((size_t)(rb * HEADS + h) * LTOK + (size_t)win * NTOK + m) * NREF;
  float v = (tid < NREF) ? src[tid] : -3.0e38f;
  __shared__ float s[128];
  s[tid] = v; __syncthreads();
  for (int st = 64; st > 0; st >>= 1) { if (tid < st) s[tid] = fmaxf(s[tid], s[tid + st]); __syncthreads(); }
  float mx = s[0]; __syncthreads();
  float e = (tid < NREF) ? __expf(v - mx) : 0.0f;
  s[tid] = e; __syncthreads();
  for (int st = 64; st > 0; st >>= 1) { if (tid < st) s[tid] += s[tid + st]; __syncthreads(); }
  float inv = 1.0f / s[0];
  out[(size_t)row * NREFP + tid] = f2bf((tid < NREF) ? e * inv : 0.0f);
}

__global__ __launch_bounds__(64) void softmax_attn_kernel(const float* __restrict__ attnf, u16* __restrict__ out) {
  int row = blockIdx.x, tid = threadIdx.x;       // row = (b_*16+h)*49+m
  int m = row % NTOK; int bh = row / NTOK;
  const float* src = attnf + ((size_t)bh * NTOKP + m) * NTOKP;
  float v = (tid < NTOK) ? src[tid] : -3.0e38f;
  __shared__ float s[64];
  s[tid] = v; __syncthreads();
  for (int st = 32; st > 0; st >>= 1) { if (tid < st) s[tid] = fmaxf(s[tid], s[tid + st]); __syncthreads(); }
  float mx = s[0]; __syncthreads();
  float e = (tid < NTOK) ? __expf(v - mx) : 0.0f;
  s[tid] = e; __syncthreads();
  for (int st = 32; st > 0; st >>= 1) { if (tid < st) s[tid] += s[tid + st]; __syncthreads(); }
  float inv = 1.0f / s[0];
  out[(size_t)row * NTOKP + tid] = f2bf((tid < NTOK) ? e * inv : 0.0f);
}

// stage v^T (per b_,h: HD x 64, cols >=49 garbage; attn pad cols are zero so they contribute 0)
__global__ __launch_bounds__(256) void vt_kernel(const u16* __restrict__ qkv, u16* __restrict__ vT) {
  int idx = blockIdx.x * 256 + threadIdx.x;      // 512*16*32*49
  int m = idx % NTOK;
  int d = (idx / NTOK) % HD;
  int h = (idx / (NTOK * HD)) % HEADS;
  int b_ = idx / (NTOK * HD * HEADS);
  vT[((size_t)(b_ * HEADS + h) * HD + d) * NTOKP + m] =
      qkv[((size_t)b_ * NTOK + m) * 1536 + 2 * CDIM + h * HD + d];
}

// ---------------- host launcher ----------------
extern "C" void kernel_launch(void* const* d_in, const int* in_sizes, int n_in,
                              void* d_out, int out_size, void* d_ws, size_t ws_size,
                              hipStream_t stream) {
  const float* x       = (const float*)d_in[0];
  const float* x_ref   = (const float*)d_in[1];
  const float* maskp   = (const float*)d_in[2];
  const float* n1w     = (const float*)d_in[3];
  const float* n1b     = (const float*)d_in[4];
  const float* qkv_w   = (const float*)d_in[5];
  const float* qkv_b   = (const float*)d_in[6];
  const float* diff_mu = (const float*)d_in[7];
  const float* diff_ls = (const float*)d_in[8];
  const float* rpb     = (const float*)d_in[9];
  const float* rqk_w   = (const float*)d_in[10];
  const float* rqk_b   = (const float*)d_in[11];
  const float* conv_w  = (const float*)d_in[12];
  const float* conv_b  = (const float*)d_in[13];
  const float* proj_w  = (const float*)d_in[14];
  const float* proj_b  = (const float*)d_in[15];
  const float* n2w     = (const float*)d_in[16];
  const float* n2b     = (const float*)d_in[17];
  const float* fc1_w   = (const float*)d_in[18];
  const float* fc1_b   = (const float*)d_in[19];
  const float* fc2_w   = (const float*)d_in[20];
  const float* fc2_b   = (const float*)d_in[21];
  float* out = (float*)d_out;
  char* ws = (char*)d_ws;

  u16*   xw_bf   = (u16*)(ws + OFF_XW);
  u16*   qkv_bf  = (u16*)(ws + OFF_QKV);
  float* refqk   = (float*)(ws + OFF_REFQK);
  u16*   refq_bf = (u16*)(ws + OFF_REFQ);
  u16*   refvT   = (u16*)(ws + OFF_REFVT);
  u16*   xref_bf = (u16*)(ws + OFF_XREF);
  float* stats   = (float*)(ws + OFF_STATS);
  float* ra_f    = (float*)(ws + OFF_RA);
  float* attn_f  = (float*)(ws + OFF_ATTNF);
  u16*   attn_bf = (u16*)(ws + OFF_ATTNB);
  float* u_f     = (float*)(ws + OFF_U);
  u16*   ra_bf   = (u16*)(ws + OFF_RABF);
  u16*   h1_bf   = (u16*)(ws + OFF_H1);
  u16*   qnew_bf = (u16*)(ws + OFF_QNEW);
  u16*   vT_bf   = (u16*)(ws + OFF_VT);
  u16*   aout_bf = (u16*)(ws + OFF_XW);   // aliases xw_bf (dead after QKV gemm)
  u16*   h_bf    = (u16*)(ws + OFF_XW);   // aliases attnout (dead after proj gemm)
  u16*   wqkv    = (u16*)(ws + OFF_WQKV);
  u16*   wrqk    = (u16*)(ws + OFF_WRQK);
  u16*   wproj   = (u16*)(ws + OFF_WPROJ);
  u16*   wfc1    = (u16*)(ws + OFF_WFC1);
  u16*   wfc2    = (u16*)(ws + OFF_WFC2);

  // 0. one-time weight conversions to bf16 (B^T layout is native: weights are (out,in))
  cvt_bf16_kernel<<<(1536 * 512) / 256, 256, 0, stream>>>(qkv_w, wqkv, 1536 * 512);
  cvt_bf16_kernel<<<(1024 * 512) / 256, 256, 0, stream>>>(rqk_w, wrqk, 1024 * 512);
  cvt_bf16_kernel<<<(512 * 512) / 256, 256, 0, stream>>>(proj_w, wproj, 512 * 512);
  cvt_bf16_kernel<<<(2048 * 512) / 256, 256, 0, stream>>>(fc1_w, wfc1, 2048 * 512);
  cvt_bf16_kernel<<<(512 * 2048) / 256, 256, 0, stream>>>(fc2_w, wfc2, 512 * 2048);
  // 1. LN1 + shift + window partition -> bf16
  ln_win_kernel<<<MROWS, 256, 0, stream>>>(x, n1w, n1b, xw_bf);
  // 2. x_ref -> bf16
  cvt_bf16_kernel<<<(200 * CDIM) / 256, 256, 0, stream>>>(x_ref, xref_bf, 200 * CDIM);
  // 3. QKV GEMM  (25088 x 1536 x 512)
  gemm_bf16_kernel<GM_BF16, MROWS, 1536, 512><<<dim3(1536 / BN, MROWS / BM), 256, 0, stream>>>(
      xw_bf, wqkv, qkv_b, qkv_bf, nullptr, nullptr);
  // 4. ref_qk GEMM (200 x 1024 x 512)
  gemm_bf16_kernel<GM_F32, 200, 1024, 512><<<dim3(1024 / BN, 2), 256, 0, stream>>>(
      xref_bf, wrqk, rqk_b, nullptr, refqk, nullptr);
  // 5. build ref_q / ref_v^T
  build_refs_kernel<<<(BATCH * NREF * CDIM) / 256, 256, 0, stream>>>(refqk, diff_mu, diff_ls, refq_bf, refvT);
  // 6. ra = q @ ref_k^T  (conv layout)
  ra_gemm_kernel<<<BWIN * HEADS * 4 / 8, 256, 0, stream>>>(qkv_bf, refq_bf, ra_f);
  // 7. three rounds of conv diffusion
  int conv_blocks = BATCH * HEADS * LTOK * NREF / 256;
  for (int round = 0; round < 3; ++round) {
    stats_zero_kernel<<<1, 64, 0, stream>>>(stats);
    conv_kernel<<<conv_blocks, 256, 0, stream>>>(ra_f, conv_w, conv_b, u_f, stats);
    conv_update_kernel<<<conv_blocks, 256, 0, stream>>>(u_f, stats, ra_f);
  }
  // 8. softmax over NREF -> ra_bf (padded K=128, zeros)
  softmax_ra_kernel<<<BWIN * HEADS * NTOK, 128, 0, stream>>>(ra_f, ra_bf);
  // 9. q_new = ra @ ref_v * SCALE
  qnew_gemm_kernel<<<BWIN * HEADS * 4 / 8, 256, 0, stream>>>(ra_bf, refvT, qnew_bf);
  // 10. stage v^T
  vt_kernel<<<BWIN * HEADS * HD * NTOK / 256, 256, 0, stream>>>(qkv_bf, vT_bf);
  // 11. attn = q_new @ k^T + rpb + mask (padded 64x64 logits)
  attn_gemm_kernel<<<BWIN * HEADS * 4 / 8, 256, 0, stream>>>(qnew_bf, qkv_bf, rpb, maskp, attn_f);
  // 12. softmax over 49 -> attn_bf (padded K=64, zeros)
  softmax_attn_kernel<<<BWIN * HEADS * NTOK, 64, 0, stream>>>(attn_f, attn_bf);
  // 13. out = attn @ v  -> (B_, N, C) bf16
  out_gemm_kernel<<<BWIN * HEADS * 4 / 8, 256, 0, stream>>>(attn_bf, vT_bf, aout_bf);
  // 14. proj GEMM + window reverse + residual -> d_out = x + attn_out
  gemm_bf16_kernel<GM_PROJ, MROWS, CDIM, 512><<<dim3(CDIM / BN, MROWS / BM), 256, 0, stream>>>(
      aout_bf, wproj, proj_b, nullptr, out, x);
  // 15. LN2 on d_out -> h_bf
  ln_plain_kernel<<<MROWS, 256, 0, stream>>>(out, n2w, n2b, h_bf);
  // 16. fc1 + GELU (25088 x 2048 x 512)
  gemm_bf16_kernel<GM_GELU, MROWS, 2048, 512><<<dim3(2048 / BN, MROWS / BM), 256, 0, stream>>>(
      h_bf, wfc1, fc1_b, h1_bf, nullptr, nullptr);
  // 17. fc2, add into d_out (25088 x 512 x 2048)
  gemm_bf16_kernel<GM_ADD, MROWS, CDIM, 2048><<<dim3(CDIM / BN, MROWS / BM), 256, 0, stream>>>(
      h1_bf, wfc2, fc2_b, nullptr, out, nullptr);

  (void)in_sizes; (void)n_in; (void)out_size; (void)ws_size;
}
